// HGTEmotionRecognizer_63299228008640
// MI455X (gfx1250) — matrix operand order repaired and checked by reading, not compiled, not run on hardware
//
#include <hip/hip_runtime.h>

typedef __attribute__((ext_vector_type(16))) _Float16 v16h;
typedef __attribute__((ext_vector_type(8)))  float    v8f;
typedef __attribute__((ext_vector_type(4)))  unsigned u32x4;
typedef __attribute__((ext_vector_type(8)))  int      i32x8;
typedef __attribute__((ext_vector_type(4)))  int      i32x4;

#define N_NODES 8192
#define D_DLG   64
#define L_SEQ   128
#define H_DIM   512
#define HEADS   8
#define DH      64
#define T_TYPES 3
#define R_ET    6
#define E_EDGES 32768
#define FF_DIM  2048
#define CIN_DIM 1536
#define CH_DIM  768
#define OUT_DIM 7
#define QKV_STR 1536

// ---------------------------------------------------------------- WMMA helpers
__device__ inline v8f zero8() {
    v8f z = {0.f,0.f,0.f,0.f,0.f,0.f,0.f,0.f};
    return z;
}

__device__ inline v8f wmma16(v16h a, v16h b, v8f c) {
    return __builtin_amdgcn_wmma_f32_16x16x32_f16(false, a, false, b,
                                                  (short)0, c, false, false);
}

// A fragment (16x32 f16), row-major source with stride ld (halves).
// Wave32 layout: lane&15 = row; lanes 0-15 K in {0..7,16..23}; lanes 16-31 +8.
__device__ inline v16h load_a_frag(const _Float16* p, int ld, int lane) {
    int row = lane & 15;
    int kb  = (lane >> 4) << 3;
    v16h a;
#pragma unroll
    for (int i = 0; i < 16; ++i) {
        int k = kb + i + ((i >> 3) << 3);
        a[i] = p[row * ld + k];
    }
    return a;
}

// B fragment (32x16) from a TRANSPOSED tile Bt[N][K] with col stride ld.
// B layout: lane&15 = col; lanes 0-15 K in {0..7,16..23}; lanes 16-31 +8,
// i.e. two 8-half runs of Bt row 'col' -> 2x ds_load_b128 per lane.
__device__ inline v16h load_bt_frag(const _Float16* p, int ld, int lane) {
    int col = lane & 15;
    int kb  = (lane >> 4) << 3;
    const _Float16* q = p + col * ld + kb;
    v16h b;
#pragma unroll
    for (int i = 0; i < 16; ++i) b[i] = q[i + ((i >> 3) << 3)];
    return b;
}

// C/D layout: lane&15 = col; VGPR r -> row r + 8*(lane>=16)
__device__ inline void store_c_f32(float* p, int ld, int lane, v8f c) {
    int col = lane & 15;
    int rb  = (lane >> 4) << 3;
#pragma unroll
    for (int r = 0; r < 8; ++r) p[(rb + r) * ld + col] = c[r];
}

__device__ inline float wave_sum(float s) {
#pragma unroll
    for (int off = 16; off > 0; off >>= 1) s += __shfl_xor(s, off, 32);
    return s;
}

__device__ inline unsigned f2ord(float x) {
    unsigned u = __float_as_uint(x);
    return (u & 0x80000000u) ? ~u : (u | 0x80000000u);
}
__device__ inline float ord2f(unsigned u) {
    return __uint_as_float((u & 0x80000000u) ? (u & 0x7fffffffu) : ~u);
}

__device__ inline float gelu_tanh(float x) {
    float x3 = x * x * x;
    return 0.5f * x * (1.f + tanhf(0.7978845608028654f * (x + 0.044715f * x3)));
}

// ---------------------------------------------------------------- TDM helper
// 2D f16 tile load via Tensor Data Mover (CDNA5 ISA sec 8.3/8.4).
__device__ inline void tdm_load_tile_f16(unsigned lds_off_bytes,
                                         const _Float16* gsrc,
                                         unsigned tileW, unsigned tileH,
                                         unsigned tensorW, unsigned tensorH,
                                         unsigned strideW) {
    unsigned long long ga = (unsigned long long)(size_t)gsrc;
    u32x4 g0;
    g0[0] = 1u;                                     // count=1, user mode
    g0[1] = lds_off_bytes;                          // lds_addr
    g0[2] = (unsigned)ga;                           // global_addr[31:0]
    g0[3] = (unsigned)((ga >> 32) & 0x01ffffffu)    // global_addr[56:32]
            | (2u << 30);                           // type=2 ("image")
    i32x8 g1;
    g1[0] = (int)(1u << 16);                        // data_size=1 (2 bytes)
    g1[1] = (int)((tensorW & 0xffffu) << 16);       // tensor_dim0[15:0]
    g1[2] = (int)((tensorW >> 16) | ((tensorH & 0xffffu) << 16));
    g1[3] = (int)((tensorH >> 16) | ((tileW & 0xffffu) << 16));
    g1[4] = (int)(tileH & 0xffffu);                 // tile_dim1 (tile_dim2=0)
    g1[5] = (int)strideW;                           // tensor_dim0_stride[31:0]
    g1[6] = 0;
    g1[7] = 0;
    i32x4 z4; z4[0] = z4[1] = z4[2] = z4[3] = 0;
#if __has_include(<hip/amd_detail/amd_gfx1250_TDM.h>)
    i32x8 z8;
#pragma unroll
    for (int i = 0; i < 8; ++i) z8[i] = 0;
    __builtin_amdgcn_tensor_load_to_lds(g0, g1, z4, z4, z8, 0);
#else
    __builtin_amdgcn_tensor_load_to_lds(g0, g1, z4, z4, 0);
#endif
}

// ---------------------------------------------------------------- generic GEMM
// C[M,Nn] = act(A_f32[M,K] @ W + bias) where W is supplied TRANSPOSED as
// Bt_f16[Nn][K]. M%64==0, K%32==0. Bt tile (128 n x 32 k) fetched by the TDM
// straight into LDS; OOB rows (Nn tail) are hardware zero-filled.
#define GBM 64
#define GBN 128
#define GBK 32
#define GEMM_LDS_BYTES (GBM * GBK * 2 + GBN * GBK * 2)   // 12 KB

__global__ void __launch_bounds__(256)
gemm_kernel(const float* __restrict__ A, const _Float16* __restrict__ Bt,
            const float* __restrict__ bias, float* __restrict__ C,
            _Float16* __restrict__ Ch, int M, int Nn, int K, int act) {
    extern __shared__ char dynsm[];
    _Float16* As = (_Float16*)dynsm;                         // offset 0
    const unsigned BT_OFF = GBM * GBK * 2;                   // 4096
    const _Float16* Bs = (const _Float16*)(dynsm + BT_OFF);  // [128 n][32 k]

    const int tid = threadIdx.x, lane = tid & 31, w = tid >> 5;
    const int m0 = blockIdx.x * GBM, n0 = blockIdx.y * GBN;
    const int wm = (w >> 2) * 32, wn = (w & 3) * 32;

    v8f acc[2][2];
#pragma unroll
    for (int i = 0; i < 2; ++i)
#pragma unroll
        for (int j = 0; j < 2; ++j) acc[i][j] = zero8();

    for (int k0 = 0; k0 < K; k0 += GBK) {
        // Bt tile: 128x32 f16 via TDM (wave 0 issues, waits TENSORcnt)
        if (tid < 32) {
            tdm_load_tile_f16(BT_OFF, Bt + (size_t)n0 * K + k0,
                              GBK, GBN,
                              (unsigned)(K - k0), (unsigned)(Nn - n0),
                              (unsigned)K);
        }
        {   // A tile: 64x32 f32 -> f16
            int r = tid >> 2, c = (tid & 3) * 8;
            const float* src = A + (size_t)(m0 + r) * K + k0 + c;
#pragma unroll
            for (int j = 0; j < 8; ++j) As[r * GBK + c + j] = (_Float16)src[j];
            if (k0 + GBK < K) __builtin_prefetch(src + GBK, 0, 1);
        }
        if (tid < 32) __builtin_amdgcn_s_wait_tensorcnt(0);
        __syncthreads();

        v16h af[2], bf[2];
        af[0] = load_a_frag(As + (wm +  0) * GBK, GBK, lane);
        af[1] = load_a_frag(As + (wm + 16) * GBK, GBK, lane);
        bf[0] = load_bt_frag(Bs + (wn +  0) * GBK, GBK, lane);
        bf[1] = load_bt_frag(Bs + (wn + 16) * GBK, GBK, lane);
#pragma unroll
        for (int i = 0; i < 2; ++i)
#pragma unroll
            for (int j = 0; j < 2; ++j) acc[i][j] = wmma16(af[i], bf[j], acc[i][j]);
        __syncthreads();
    }

    const int col = lane & 15, rb = (lane >> 4) << 3;
#pragma unroll
    for (int i = 0; i < 2; ++i)
#pragma unroll
        for (int j = 0; j < 2; ++j) {
            int gm = m0 + wm + i * 16 + rb;
            int gn = n0 + wn + j * 16 + col;
            if (gn < Nn) {
                float bv = bias ? bias[gn] : 0.f;
#pragma unroll
                for (int r = 0; r < 8; ++r) {
                    float v = acc[i][j][r] + bv;
                    if (act == 1) v = fmaxf(v, 0.f);
                    if (Ch) Ch[(size_t)(gm + r) * Nn + gn] = (_Float16)v;
                    else     C[(size_t)(gm + r) * Nn + gn] = v;
                }
            }
        }
}

// ---------------------------------------------------------------- attention
// One block per (dialogue d, head h). Flash-style, KV blocks of 32.
// Kt staged as [kcol][e] (natural global layout), V staged transposed [vcol][e2]
// so all WMMA B fragments are lane-contiguous.
__global__ void __launch_bounds__(256)
attn_kernel(const float* __restrict__ qkv, float* __restrict__ out) {
    const int d = blockIdx.x, h = blockIdx.y;
    __shared__ _Float16 Kt[32 * 64];   // [kcol][feat e]
    __shared__ _Float16 Vt[64 * 32];   // [vcol][e2]
    __shared__ float    Sb[128 * 32];
    __shared__ _Float16 Pb[128 * 32];
    __shared__ float    fac[128];

    const int tid = threadIdx.x, lane = tid & 31, w = tid >> 5;

    v16h qf[2];
    {
        int row = lane & 15;
        int n   = d * L_SEQ + w * 16 + row;
        const float* qp = qkv + (size_t)n * QKV_STR + h * DH;
        int kb = (lane >> 4) << 3;
#pragma unroll
        for (int s = 0; s < 2; ++s)
#pragma unroll
            for (int i = 0; i < 16; ++i) {
                int e = s * 32 + kb + i + ((i >> 3) << 3);
                qf[s][i] = (_Float16)(qp[e] * 0.125f);
            }
    }

    float mrow = -3.0e38f, lrow = 0.f;
    v8f o[4];
#pragma unroll
    for (int c = 0; c < 4; ++c) o[c] = zero8();

    for (int blk = 0; blk < 4; ++blk) {
        const int kb0 = blk * 32;
        {   // K block: Kt[kcol][e]  (row-contiguous copies)
            int kcol = tid >> 3, e0 = (tid & 7) * 8;
            const float* p = qkv + (size_t)(d * L_SEQ + kb0 + kcol) * QKV_STR
                             + H_DIM + h * DH + e0;
#pragma unroll
            for (int j = 0; j < 8; ++j)
                Kt[kcol * 64 + e0 + j] = (_Float16)p[j];
        }
        {   // V block transposed: Vt[vcol][e2]
            int vcol = tid & 63, e2b = (tid >> 6) * 8;
            const float* p = qkv + (size_t)(d * L_SEQ + kb0 + e2b) * QKV_STR
                             + 2 * H_DIM + h * DH + vcol;
#pragma unroll
            for (int j = 0; j < 8; ++j)
                Vt[vcol * 32 + e2b + j] = (_Float16)p[(size_t)j * QKV_STR];
        }
        __syncthreads();

        // S strip = Q(16x64) * K^T(64x32)
#pragma unroll
        for (int ct = 0; ct < 2; ++ct) {
            v8f s = zero8();
#pragma unroll
            for (int st = 0; st < 2; ++st)
                s = wmma16(qf[st],
                           load_bt_frag(Kt + (ct * 16) * 64 + st * 32, 64, lane), s);
            store_c_f32(Sb + (16 * w) * 32 + ct * 16, 32, lane, s);
        }
        __syncthreads();

        if (tid < 128) {
            float bm = -3.0e38f;
#pragma unroll
            for (int c = 0; c < 32; ++c) bm = fmaxf(bm, Sb[tid * 32 + c]);
            float nm = fmaxf(mrow, bm);
            float f  = __expf(mrow - nm);
            float rs = 0.f;
#pragma unroll
            for (int c = 0; c < 32; ++c) {
                float p = __expf(Sb[tid * 32 + c] - nm);
                Pb[tid * 32 + c] = (_Float16)p;
                rs += p;
            }
            lrow = lrow * f + rs;
            mrow = nm;
            fac[tid] = f;
        }
        __syncthreads();

        const int rb = (lane >> 4) << 3;
        float fr[8];
#pragma unroll
        for (int r = 0; r < 8; ++r) fr[r] = fac[16 * w + rb + r];
        v16h pa = load_a_frag(Pb + (16 * w) * 32, 32, lane);
#pragma unroll
        for (int ct = 0; ct < 4; ++ct) {
#pragma unroll
            for (int r = 0; r < 8; ++r) o[ct][r] *= fr[r];
            o[ct] = wmma16(pa, load_bt_frag(Vt + (ct * 16) * 32, 32, lane), o[ct]);
        }
        __syncthreads();
    }

    if (tid < 128) fac[tid] = 1.f / lrow;
    __syncthreads();
    const int col = lane & 15, rb = (lane >> 4) << 3;
#pragma unroll
    for (int ct = 0; ct < 4; ++ct)
#pragma unroll
        for (int r = 0; r < 8; ++r) {
            int row = 16 * w + rb + r;
            out[(size_t)(d * L_SEQ + row) * H_DIM + h * DH + ct * 16 + col] =
                o[ct][r] * fac[row];
        }
}

// ---------------------------------------------------------------- HGT edges
// Phase 1: kj = K[src] @ arel[h]; lg = <q[dst], kj> * prel * scale
// relT is transposed per head: relT[h][f][d]. LDS: Ak | BaT (TDM) | kj
#define ELG_LDS_BYTES (128 * 64 * 2 + 64 * 64 * 2 + 128 * 64 * 4)

__global__ void __launch_bounds__(256)
edge_lg_kernel(const _Float16* __restrict__ Kmat, const float* __restrict__ Qmat,
               const _Float16* __restrict__ arelT, const float* __restrict__ prel,
               const int* __restrict__ src, const int* __restrict__ dst,
               float* __restrict__ lg) {
    extern __shared__ char dynsm[];
    _Float16* Ak = (_Float16*)dynsm;
    const unsigned BA_OFF = 128 * 64 * 2;
    const _Float16* Ba = (const _Float16*)(dynsm + BA_OFF);  // [f][d]
    float* kj = (float*)(dynsm + BA_OFF + 64 * 64 * 2);

    const int tid = threadIdx.x, lane = tid & 31, w = tid >> 5;
    const int e0 = blockIdx.x * 128, h = blockIdx.y;

    if (tid < 32)   // arelT[h] 64x64 f16 contiguous -> TDM copy
        tdm_load_tile_f16(BA_OFF, arelT + (size_t)h * 4096, 4096, 1, 4096, 1, 4096);
    {   // gather K rows (f16)
        int r = tid >> 1, c0 = (tid & 1) * 32;
        const _Float16* p = Kmat + (size_t)src[e0 + r] * H_DIM + h * DH + c0;
#pragma unroll
        for (int j = 0; j < 32; ++j) Ak[r * 64 + c0 + j] = p[j];
    }
    if (tid < 32) __builtin_amdgcn_s_wait_tensorcnt(0);
    __syncthreads();

    v16h af[2];
    af[0] = load_a_frag(Ak + (16 * w) * 64 +  0, 64, lane);
    af[1] = load_a_frag(Ak + (16 * w) * 64 + 32, 64, lane);
#pragma unroll
    for (int ct = 0; ct < 4; ++ct) {
        v8f a = zero8();
#pragma unroll
        for (int s = 0; s < 2; ++s)
            a = wmma16(af[s],
                       load_bt_frag(Ba + (ct * 16) * 64 + s * 32, 64, lane), a);
        store_c_f32(kj + (16 * w) * 64 + ct * 16, 64, lane, a);
    }
    __syncthreads();

    if (tid < 128) {
        int e = e0 + tid;
        const float* q = Qmat + (size_t)dst[e] * H_DIM + h * DH;
        float acc = 0.f;
#pragma unroll
        for (int f = 0; f < 64; ++f) acc += q[f] * kj[tid * 64 + f];
        lg[(size_t)e * HEADS + h] = acc * prel[h] * 0.125f;
    }
}

__global__ void __launch_bounds__(256)
edge_max_kernel(const float* __restrict__ lg, const int* __restrict__ dst,
                unsigned* __restrict__ nm) {
    int idx = blockIdx.x * 256 + threadIdx.x;   // E*HEADS
    int e = idx >> 3, h = idx & 7;
    atomicMax(&nm[(size_t)dst[e] * HEADS + h], f2ord(lg[idx]));
}

__global__ void __launch_bounds__(256)
edge_exp_kernel(float* __restrict__ lg, const int* __restrict__ dst,
                const unsigned* __restrict__ nm, float* __restrict__ ns) {
    int idx = blockIdx.x * 256 + threadIdx.x;
    int e = idx >> 3, h = idx & 7;
    float m  = ord2f(nm[(size_t)dst[e] * HEADS + h]);
    float ev = __expf(lg[idx] - m);
    lg[idx]  = ev;
    atomicAdd(&ns[(size_t)dst[e] * HEADS + h], ev);
}

// Phase 2: mj = V[src] @ mrel[h]; agg[dst] += alpha * mj
#define EAG_LDS_BYTES (128 * 64 * 2 + 64 * 64 * 2)

__global__ void __launch_bounds__(256)
edge_agg_kernel(const _Float16* __restrict__ Vmat, const _Float16* __restrict__ mrelT,
                const int* __restrict__ src, const int* __restrict__ dst,
                const float* __restrict__ lgnum, const float* __restrict__ ns,
                float* __restrict__ agg) {
    extern __shared__ char dynsm[];
    _Float16* Av = (_Float16*)dynsm;
    const unsigned BM_OFF = 128 * 64 * 2;
    const _Float16* Bm = (const _Float16*)(dynsm + BM_OFF);  // [f][d]

    const int tid = threadIdx.x, lane = tid & 31, w = tid >> 5;
    const int e0 = blockIdx.x * 128, h = blockIdx.y;

    if (tid < 32)
        tdm_load_tile_f16(BM_OFF, mrelT + (size_t)h * 4096, 4096, 1, 4096, 1, 4096);
    {
        int r = tid >> 1, c0 = (tid & 1) * 32;
        const _Float16* p = Vmat + (size_t)src[e0 + r] * H_DIM + h * DH + c0;
#pragma unroll
        for (int j = 0; j < 32; ++j) Av[r * 64 + c0 + j] = p[j];
    }
    if (tid < 32) __builtin_amdgcn_s_wait_tensorcnt(0);
    __syncthreads();

    v16h af[2];
    af[0] = load_a_frag(Av + (16 * w) * 64 +  0, 64, lane);
    af[1] = load_a_frag(Av + (16 * w) * 64 + 32, 64, lane);
    v8f acc[4];
#pragma unroll
    for (int ct = 0; ct < 4; ++ct) {
        acc[ct] = zero8();
#pragma unroll
        for (int s = 0; s < 2; ++s)
            acc[ct] = wmma16(af[s],
                             load_bt_frag(Bm + (ct * 16) * 64 + s * 32, 64, lane),
                             acc[ct]);
    }

    const int col = lane & 15, rb = (lane >> 4) << 3;
#pragma unroll
    for (int r = 0; r < 8; ++r) {
        int   e  = e0 + 16 * w + rb + r;
        int   dn = dst[e];
        float al = lgnum[(size_t)e * HEADS + h] /
                   (ns[(size_t)dn * HEADS + h] + 1e-9f);
        float* ap = agg + (size_t)dn * H_DIM + h * DH;
#pragma unroll
        for (int ct = 0; ct < 4; ++ct)
            atomicAdd(ap + ct * 16 + col, acc[ct][r] * al);
    }
}

// ---------------------------------------------------------------- elementwise
__global__ void __launch_bounds__(256)
ln_residual_kernel(float* __restrict__ h, const float* __restrict__ add,
                   const float* __restrict__ g, const float* __restrict__ b) {
    const int row  = blockIdx.x * 8 + (threadIdx.x >> 5);
    const int lane = threadIdx.x & 31;
    const size_t base = (size_t)row * H_DIM;
    float x[16];
    float s = 0.f;
#pragma unroll
    for (int i = 0; i < 16; ++i) {
        int c = lane + 32 * i;
        x[i] = h[base + c] + add[base + c];
        s += x[i];
    }
    float mu = wave_sum(s) * (1.f / H_DIM);
    float v = 0.f;
#pragma unroll
    for (int i = 0; i < 16; ++i) { float d = x[i] - mu; v += d * d; }
    float rstd = rsqrtf(wave_sum(v) * (1.f / H_DIM) + 1e-5f);
#pragma unroll
    for (int i = 0; i < 16; ++i) {
        int c = lane + 32 * i;
        h[base + c] = (x[i] - mu) * rstd * g[c] + b[c];
    }
}

__global__ void __launch_bounds__(256)
hgt_post_kernel(float* __restrict__ cur, const float* __restrict__ outp,
                const float* __restrict__ g, const float* __restrict__ b,
                const float* __restrict__ skipp) {
    const int row  = blockIdx.x * 8 + (threadIdx.x >> 5);
    const int lane = threadIdx.x & 31;
    const float beta = 1.f / (1.f + __expf(-skipp[0]));
    const size_t base = (size_t)row * H_DIM;
    float x[16];
    float s = 0.f;
#pragma unroll
    for (int i = 0; i < 16; ++i) {
        int c = lane + 32 * i;
        x[i] = beta * outp[base + c] + (1.f - beta) * cur[base + c];
        s += x[i];
    }
    float mu = wave_sum(s) * (1.f / H_DIM);
    float v = 0.f;
#pragma unroll
    for (int i = 0; i < 16; ++i) { float d = x[i] - mu; v += d * d; }
    float rstd = rsqrtf(wave_sum(v) * (1.f / H_DIM) + 1e-5f);
#pragma unroll
    for (int i = 0; i < 16; ++i) {
        int c = lane + 32 * i;
        cur[base + c] = fmaxf((x[i] - mu) * rstd * g[c] + b[c], 0.f);
    }
}

__global__ void __launch_bounds__(256)
gelu_kernel(float* __restrict__ p, unsigned n) {
    unsigned i = blockIdx.x * 256u + threadIdx.x;
    if (i < n) p[i] = gelu_tanh(p[i]);
}

__global__ void __launch_bounds__(256)
xcat_kernel(float* __restrict__ dst, const float* __restrict__ x,
            const float* __restrict__ spk_emb, const int* __restrict__ spk_idx) {
    unsigned i = blockIdx.x * 256u + threadIdx.x;   // N*576
    if (i >= (unsigned)N_NODES * 576u) return;
    unsigned n = i / 576u, c = i % 576u;
    dst[i] = (c < 512u) ? x[(size_t)n * 512 + c]
                        : spk_emb[(size_t)spk_idx[n] * 64 + (c - 512u)];
}

__global__ void __launch_bounds__(256)
copy_cols_kernel(float* __restrict__ dst, int dstStride,
                 const float* __restrict__ src) {
    unsigned i = blockIdx.x * 256u + threadIdx.x;   // N*512
    if (i >= (unsigned)N_NODES * 512u) return;
    unsigned n = i / 512u, c = i % 512u;
    dst[(size_t)n * dstStride + c] = src[i];
}

__global__ void __launch_bounds__(256)
fill_f32_kernel(float* __restrict__ p, float v, unsigned n) {
    unsigned i = blockIdx.x * 256u + threadIdx.x;
    if (i < n) p[i] = v;
}

__global__ void __launch_bounds__(256)
init_stats_kernel(unsigned* __restrict__ nm, float* __restrict__ ns, unsigned n) {
    unsigned i = blockIdx.x * 256u + threadIdx.x;
    if (i < n) { nm[i] = f2ord(-3.0e38f); ns[i] = 0.f; }
}

// Batched f32 [K][N] -> f16 TRANSPOSED [N][K]; batch via blockIdx.y.
__global__ void __launch_bounds__(256)
cvt_t_kernel(const float* __restrict__ src, _Float16* __restrict__ dst,
             int K, int N) {
    unsigned i = blockIdx.x * 256u + threadIdx.x;
    unsigned total = (unsigned)K * (unsigned)N;
    if (i >= total) return;
    size_t b = (size_t)blockIdx.y * total;
    unsigned k = i / (unsigned)N, n = i % (unsigned)N;
    dst[b + (size_t)n * K + k] = (_Float16)src[b + i];
}

// ---------------------------------------------------------------- host driver
static inline void launch_gemm(const float* A, const _Float16* Bt, const float* bias,
                               float* C, _Float16* Ch, int M, int Nn, int K, int act,
                               hipStream_t s) {
    dim3 grid(M / GBM, (Nn + GBN - 1) / GBN);
    gemm_kernel<<<grid, dim3(256), GEMM_LDS_BYTES, s>>>(A, Bt, bias, C, Ch, M, Nn, K, act);
}

extern "C" void kernel_launch(void* const* d_in, const int* in_sizes, int n_in,
                              void* d_out, int out_size, void* d_ws, size_t ws_size,
                              hipStream_t stream) {
    (void)in_sizes; (void)n_in; (void)out_size; (void)ws_size;

    const float* x_in[3]   = {(const float*)d_in[0], (const float*)d_in[1],
                              (const float*)d_in[2]};
    const float* spk_emb   = (const float*)d_in[3];
    const float* proj_w    = (const float*)d_in[4];
    const float* proj_b    = (const float*)d_in[5];
    const float* t_qkv_w   = (const float*)d_in[6];
    const float* t_qkv_b   = (const float*)d_in[7];
    const float* t_out_w   = (const float*)d_in[8];
    const float* t_out_b   = (const float*)d_in[9];
    const float* t_ff1_w   = (const float*)d_in[10];
    const float* t_ff1_b   = (const float*)d_in[11];
    const float* t_ff2_w   = (const float*)d_in[12];
    const float* t_ff2_b   = (const float*)d_in[13];
    const float* t_ln1_g   = (const float*)d_in[14];
    const float* t_ln1_b   = (const float*)d_in[15];
    const float* t_ln2_g   = (const float*)d_in[16];
    const float* t_ln2_b   = (const float*)d_in[17];
    const float* g_k_w     = (const float*)d_in[18];
    const float* g_k_b     = (const float*)d_in[19];
    const float* g_q_w     = (const float*)d_in[20];
    const float* g_q_b     = (const float*)d_in[21];
    const float* g_v_w     = (const float*)d_in[22];
    const float* g_v_b     = (const float*)d_in[23];
    const float* g_a_w     = (const float*)d_in[24];
    const float* g_a_b     = (const float*)d_in[25];
    const float* g_skip    = (const float*)d_in[26];
    const float* g_arel    = (const float*)d_in[27];
    const float* g_mrel    = (const float*)d_in[28];
    const float* g_prel    = (const float*)d_in[29];
    const float* g_ln_g    = (const float*)d_in[30];
    const float* g_ln_b    = (const float*)d_in[31];
    const float* c1_w      = (const float*)d_in[32];
    const float* c1_b      = (const float*)d_in[33];
    const float* c2_w      = (const float*)d_in[34];
    const float* c2_b      = (const float*)d_in[35];
    const int*   spk_idx   = (const int*)d_in[36];
    const int*   edge_idx  = (const int*)d_in[38];

    const int SRC_T[R_ET] = {0, 1, 0, 2, 1, 2};
    const int DST_T[R_ET] = {1, 0, 2, 0, 2, 1};

    const size_t NH = (size_t)N_NODES * H_DIM;
    float* W = (float*)d_ws;
    float* cur[3] = {W, W + NH, W + 2 * NH};
    float* bufA  = W + 3 * NH;                            // 3*N*576 : xcat / Qp
    float* bufB  = bufA + (size_t)3 * N_NODES * 576;      // 4*NH : qkv/ff/agg/ci
    float* bufC  = bufB + 4 * NH;                         // 2*NH : attn/o/clsh
    float* lgbuf = bufC + 2 * NH;                         // R*E*HEADS
    unsigned* nodemax = (unsigned*)(lgbuf + (size_t)R_ET * E_EDGES * HEADS);
    float* nodesum = (float*)(nodemax + (size_t)T_TYPES * N_NODES * HEADS);
    _Float16* hptr = (_Float16*)(nodesum + (size_t)T_TYPES * N_NODES * HEADS);

    // --- one-time f32 -> f16 TRANSPOSED weight conversion (WMMA B operands)
    auto carve_cvt_t = [&](const float* src, int batch, int K, int N) -> const _Float16* {
        _Float16* p = hptr;
        size_t n = (size_t)batch * K * N;
        hptr += n;
        unsigned kn = (unsigned)K * (unsigned)N;
        dim3 grid((kn + 255u) / 256u, batch);
        cvt_t_kernel<<<grid, 256, 0, stream>>>(src, p, K, N);
        return p;
    };
    const _Float16* proj_wt  = carve_cvt_t(proj_w,  T_TYPES, 576, H_DIM);
    const _Float16* t_qkv_wt = carve_cvt_t(t_qkv_w, 6, H_DIM, QKV_STR);
    const _Float16* t_out_wt = carve_cvt_t(t_out_w, 6, H_DIM, H_DIM);
    const _Float16* t_ff1_wt = carve_cvt_t(t_ff1_w, 6, H_DIM, FF_DIM);
    const _Float16* t_ff2_wt = carve_cvt_t(t_ff2_w, 6, FF_DIM, H_DIM);
    const _Float16* g_k_wt   = carve_cvt_t(g_k_w,   6, H_DIM, H_DIM);
    const _Float16* g_q_wt   = carve_cvt_t(g_q_w,   6, H_DIM, H_DIM);
    const _Float16* g_v_wt   = carve_cvt_t(g_v_w,   6, H_DIM, H_DIM);
    const _Float16* g_a_wt   = carve_cvt_t(g_a_w,   6, H_DIM, H_DIM);
    const _Float16* c1_wt    = carve_cvt_t(c1_w,    1, CIN_DIM, CH_DIM);
    const _Float16* c2_wt    = carve_cvt_t(c2_w,    1, CH_DIM, OUT_DIM);
    const _Float16* g_arelt  = carve_cvt_t(g_arel,  2 * R_ET * HEADS, DH, DH);
    const _Float16* g_mrelt  = carve_cvt_t(g_mrel,  2 * R_ET * HEADS, DH, DH);
    // f16 K/V projection buffers for the HGT edge kernels
    _Float16* Kh[3]; _Float16* Vh[3];
    for (int t = 0; t < T_TYPES; ++t) { Kh[t] = hptr; hptr += NH;
                                        Vh[t] = hptr; hptr += NH; }

    // --- input projection: xcat = [x_t | spk], cur_t = xcat @ proj_w + b
    for (int t = 0; t < T_TYPES; ++t) {
        float* xcat = bufA + (size_t)t * N_NODES * 576;
        xcat_kernel<<<(N_NODES * 576 + 255) / 256, 256, 0, stream>>>(
            xcat, x_in[t], spk_emb, spk_idx);
        launch_gemm(xcat, proj_wt + (size_t)t * 576 * H_DIM, proj_b + t * H_DIM,
                    cur[t], nullptr, N_NODES, H_DIM, 576, 0, stream);
    }

    float* attn_out = bufC;
    float* obuf     = bufC + NH;

    // --- per-type transformer encoders
    for (int t = 0; t < T_TYPES; ++t) {
        for (int l = 0; l < 2; ++l) {
            const size_t wl = (size_t)(t * 2 + l);
            float* qkv = bufB;
            float* ff  = bufB;
            launch_gemm(cur[t], t_qkv_wt + wl * H_DIM * QKV_STR,
                        t_qkv_b + wl * QKV_STR, qkv, nullptr,
                        N_NODES, QKV_STR, H_DIM, 0, stream);
            attn_kernel<<<dim3(D_DLG, HEADS), 256, 0, stream>>>(qkv, attn_out);
            launch_gemm(attn_out, t_out_wt + wl * H_DIM * H_DIM,
                        t_out_b + wl * H_DIM, obuf, nullptr,
                        N_NODES, H_DIM, H_DIM, 0, stream);
            ln_residual_kernel<<<N_NODES / 8, 256, 0, stream>>>(
                cur[t], obuf, t_ln1_g + wl * H_DIM, t_ln1_b + wl * H_DIM);
            launch_gemm(cur[t], t_ff1_wt + wl * H_DIM * FF_DIM,
                        t_ff1_b + wl * FF_DIM, ff, nullptr,
                        N_NODES, FF_DIM, H_DIM, 1, stream);
            launch_gemm(ff, t_ff2_wt + wl * FF_DIM * H_DIM,
                        t_ff2_b + wl * H_DIM, obuf, nullptr,
                        N_NODES, H_DIM, FF_DIM, 0, stream);
            ln_residual_kernel<<<N_NODES / 8, 256, 0, stream>>>(
                cur[t], obuf, t_ln2_g + wl * H_DIM, t_ln2_b + wl * H_DIM);
        }
    }

    // --- HGT layers
    for (int l = 0; l < 2; ++l) {
        float* Qp[3];
        for (int t = 0; t < T_TYPES; ++t) {
            const size_t wl = (size_t)(l * T_TYPES + t);
            Qp[t] = bufA + (size_t)t * NH;
            launch_gemm(cur[t], g_k_wt + wl * H_DIM * H_DIM, g_k_b + wl * H_DIM,
                        nullptr, Kh[t], N_NODES, H_DIM, H_DIM, 0, stream);
            launch_gemm(cur[t], g_q_wt + wl * H_DIM * H_DIM, g_q_b + wl * H_DIM,
                        Qp[t], nullptr, N_NODES, H_DIM, H_DIM, 0, stream);
            launch_gemm(cur[t], g_v_wt + wl * H_DIM * H_DIM, g_v_b + wl * H_DIM,
                        nullptr, Vh[t], N_NODES, H_DIM, H_DIM, 0, stream);
        }
        float* agg = bufB;   // [3][N][H]
        {
            unsigned ns = T_TYPES * N_NODES * HEADS;
            init_stats_kernel<<<(ns + 255) / 256, 256, 0, stream>>>(nodemax, nodesum, ns);
            unsigned na = (unsigned)(3 * NH);
            fill_f32_kernel<<<(na + 255) / 256, 256, 0, stream>>>(agg, 0.f, na);
        }
        for (int r = 0; r < R_ET; ++r) {
            const int st = SRC_T[r], dt = DST_T[r];
            const int* srcp = edge_idx + (size_t)r * 2 * E_EDGES;
            const int* dstp = srcp + E_EDGES;
            edge_lg_kernel<<<dim3(E_EDGES / 128, HEADS), 256, ELG_LDS_BYTES, stream>>>(
                Kh[st], Qp[dt],
                g_arelt + (size_t)(l * R_ET + r) * HEADS * DH * DH,
                g_prel + (size_t)(l * R_ET + r) * HEADS,
                srcp, dstp, lgbuf + (size_t)r * E_EDGES * HEADS);
        }
        for (int r = 0; r < R_ET; ++r) {
            const int dt = DST_T[r];
            const int* dstp = edge_idx + (size_t)r * 2 * E_EDGES + E_EDGES;
            edge_max_kernel<<<(E_EDGES * HEADS) / 256, 256, 0, stream>>>(
                lgbuf + (size_t)r * E_EDGES * HEADS, dstp,
                nodemax + (size_t)dt * N_NODES * HEADS);
        }
        for (int r = 0; r < R_ET; ++r) {
            const int dt = DST_T[r];
            const int* dstp = edge_idx + (size_t)r * 2 * E_EDGES + E_EDGES;
            edge_exp_kernel<<<(E_EDGES * HEADS) / 256, 256, 0, stream>>>(
                lgbuf + (size_t)r * E_EDGES * HEADS, dstp,
                nodemax + (size_t)dt * N_NODES * HEADS,
                nodesum + (size_t)dt * N_NODES * HEADS);
        }
        for (int r = 0; r < R_ET; ++r) {
            const int st = SRC_T[r], dt = DST_T[r];
            const int* srcp = edge_idx + (size_t)r * 2 * E_EDGES;
            const int* dstp = srcp + E_EDGES;
            edge_agg_kernel<<<dim3(E_EDGES / 128, HEADS), 256, EAG_LDS_BYTES, stream>>>(
                Vh[st], g_mrelt + (size_t)(l * R_ET + r) * HEADS * DH * DH,
                srcp, dstp, lgbuf + (size_t)r * E_EDGES * HEADS,
                nodesum + (size_t)dt * N_NODES * HEADS,
                agg + (size_t)dt * NH);
        }
        {
            unsigned na = (unsigned)(3 * NH);
            gelu_kernel<<<(na + 255) / 256, 256, 0, stream>>>(agg, na);
        }
        for (int t = 0; t < T_TYPES; ++t) {
            const size_t wl = (size_t)(l * T_TYPES + t);
            launch_gemm(agg + (size_t)t * NH, g_a_wt + wl * H_DIM * H_DIM,
                        g_a_b + wl * H_DIM, obuf, nullptr,
                        N_NODES, H_DIM, H_DIM, 0, stream);
            hgt_post_kernel<<<N_NODES / 8, 256, 0, stream>>>(
                cur[t], obuf, g_ln_g + wl * H_DIM, g_ln_b + wl * H_DIM,
                g_skip + wl);
        }
    }

    // --- classifier
    float* ci   = bufB;          // [N, 1536]
    float* clsh = bufC;          // [N, 768]
    for (int t = 0; t < T_TYPES; ++t)
        copy_cols_kernel<<<(N_NODES * 512 + 255) / 256, 256, 0, stream>>>(
            ci + t * H_DIM, CIN_DIM, cur[t]);
    launch_gemm(ci, c1_wt, c1_b, clsh, nullptr, N_NODES, CH_DIM, CIN_DIM, 1, stream);
    launch_gemm(clsh, c2_wt, c2_b, (float*)d_out, nullptr,
                N_NODES, OUT_DIM, CH_DIM, 0, stream);
}
